// UniversalTransformerBlock_33681133535742
// MI455X (gfx1250) — matrix operand order repaired and checked by reading, not compiled, not run on hardware
//
#include <hip/hip_runtime.h>
#include <hip/hip_bf16.h>
#include <math.h>

// ---------------------------------------------------------------------------
// Types for CDNA5 WMMA (wave32): v_wmma_f32_16x16x32_bf16
// ---------------------------------------------------------------------------
typedef __attribute__((ext_vector_type(16))) __bf16 v16bf;
typedef __attribute__((ext_vector_type(8)))  __bf16 bf16x8;
typedef __attribute__((ext_vector_type(8)))  float  v8f;
typedef __attribute__((ext_vector_type(4)))  unsigned int v4u;
typedef __attribute__((ext_vector_type(8)))  unsigned int v8u;

#define D_MODEL 1024
#define N_HEADS 16
#define HEAD_DIM 64
#define D_FF 4096
#define SEQ 1024
#define BATCH 8
#define BSEQ (BATCH * SEQ)   // 8192 rows

__device__ __forceinline__ v16bf cat8(bf16x8 lo, bf16x8 hi) {
  return __builtin_shufflevector(lo, hi, 0, 1, 2, 3, 4, 5, 6, 7,
                                         8, 9, 10, 11, 12, 13, 14, 15);
}

__device__ __forceinline__ v8f wmma_bf16(v16bf a, v16bf b, v8f c) {
  // 8-arg form: (neg_a, A, neg_b, B, c_mod, C, reuse_a, reuse_b)
  return __builtin_amdgcn_wmma_f32_16x16x32_bf16(false, a, false, b,
                                                 (short)0, c, false, false);
}

#if __has_builtin(__builtin_amdgcn_s_wait_tensorcnt)
#define TDM_WAIT(n) __builtin_amdgcn_s_wait_tensorcnt(n)
#else
#define TDM_WAIT(n) asm volatile("s_wait_tensorcnt %0" ::"i"(n))
#endif

// ---------------------------------------------------------------------------
// Tensor Data Mover: issue a 2D tile load Global->LDS (one instruction per
// wave; EXEC ignored; tracked with TENSORcnt).  D# per CDNA5 ISA:
//   group0: [1:0]=count=1, [63:32]=lds_addr, [120:64]=global_addr, type=2
//   group1: data_size=2B, pad (row 32 DWORDs + 4 DWORDs), tensor/tile dims,
//           dim0 stride (elements)
// Tile: tile_d0 elems contiguous (dim0) x tile_d1 rows, row stride = stride0.
// LDS rows padded: interval code 4 (32 DWORDs = 64 elems), amount code 3
// (4 DWORDs = 8 elems) -> LDS row stride 72 bf16.
// ---------------------------------------------------------------------------
__device__ __forceinline__ void tdm_load_2d(unsigned lds_addr, const void* gptr,
                                            unsigned tile_d0, unsigned tile_d1,
                                            unsigned tensor_d0, unsigned tensor_d1,
                                            unsigned long stride0) {
  const unsigned long ga = (unsigned long)(uintptr_t)gptr;
  v4u g0;
  g0[0] = 1u;                                       // count=1, user descriptor
  g0[1] = lds_addr;                                 // LDS byte address
  g0[2] = (unsigned)(ga & 0xFFFFFFFFu);             // global_addr[31:0]
  g0[3] = (unsigned)((ga >> 32) & 0x1FFFFFFu) | (2u << 30);  // addr[56:32]|type=2
  v8u g1;
  g1[0] = (1u << 16)                                // data_size = 1 (2 bytes)
        | (1u << 20)                                // pad_enable
        | (4u << 22)                                // pad_interval: 32 DWORDs
        | (3u << 25);                               // pad_amount: 4 DWORDs
  g1[1] = (tensor_d0 & 0xFFFFu) << 16;              // tensor_dim0[15:0]
  g1[2] = ((tensor_d0 >> 16) & 0xFFFFu)             // tensor_dim0[31:16]
        | ((tensor_d1 & 0xFFFFu) << 16);            // tensor_dim1[15:0]
  g1[3] = ((tensor_d1 >> 16) & 0xFFFFu)             // tensor_dim1[31:16]
        | (tile_d0 << 16);                          // tile_dim0
  g1[4] = tile_d1;                                  // tile_dim1 | tile_dim2=0
  g1[5] = (unsigned)(stride0 & 0xFFFFFFFFu);        // dim0_stride[31:0]
  g1[6] = (unsigned)((stride0 >> 32) & 0xFFFFu);    // dim0_stride[47:32]
  g1[7] = 0u;                                       // dim1_stride unused (2D)
  asm volatile("tensor_load_to_lds %0, %1" ::"s"(g0), "s"(g1) : "memory");
}

__device__ __forceinline__ unsigned lds_off(const void* p) {
  return (unsigned)(uintptr_t)p;  // low 32 bits of a generic LDS pointer
}

// ---------------------------------------------------------------------------
// Kernel 1: x += step_embed[step]; emit f32 residual + bf16 GEMM operand
// ---------------------------------------------------------------------------
__global__ void prep_kernel(const float* __restrict__ x,
                            const float* __restrict__ se,
                            const int* __restrict__ stepp,
                            float* __restrict__ xres,
                            __bf16* __restrict__ xb, int n) {
  int i = blockIdx.x * blockDim.x + threadIdx.x;
  if (i >= n) return;
  int step = stepp[0];
  step = step > 15 ? 15 : (step < 0 ? 0 : step);
  float v = x[i] + se[step * D_MODEL + (i & (D_MODEL - 1))];
  xres[i] = v;
  xb[i] = (__bf16)v;
}

// ---------------------------------------------------------------------------
// Kernel 2: weight convert f32 [K][N] -> bf16 [N][K] (transposed for B-frags)
// ---------------------------------------------------------------------------
__global__ void wt_convert_kernel(const float* __restrict__ w,
                                  __bf16* __restrict__ wt, int K, int N) {
  long i = (long)blockIdx.x * blockDim.x + threadIdx.x;
  long total = (long)K * N;
  if (i >= total) return;
  long n = i / K, k = i % K;
  wt[i] = (__bf16)w[k * (long)N + n];
}

// ---------------------------------------------------------------------------
// Kernel 3: bf16 WMMA GEMM  C[M][N] = A[M][K] * Bt[N][K]^T + bias.
//   EPI=0: f32 out    EPI=1: bf16 out    EPI=2: exact GELU then bf16 out
// Block = 256 threads = 8 waves (2x4), tile 128x128x64.
// Tiles staged by the Tensor Data Mover (wave 0 issues tensor_load_to_lds),
// double-buffered so DMA of tile k+1 overlaps WMMA on tile k.
// ---------------------------------------------------------------------------
template <int EPI>
__global__ __launch_bounds__(256)
void gemm_bf16_kernel(const __bf16* __restrict__ A, const __bf16* __restrict__ Bt,
                      const float* __restrict__ bias, float* __restrict__ Cf,
                      __bf16* __restrict__ Cb, int M, int N, int K) {
  constexpr int BM = 128, BN = 128, BK = 64, LDSS = 72;  // 64 elems + 8 pad
  __shared__ __align__(16) __bf16 sA[2][BM][LDSS];
  __shared__ __align__(16) __bf16 sB[2][BN][LDSS];

  const int tid = threadIdx.x;
  const int wave = tid >> 5, lane = tid & 31;
  const int wm = wave >> 2, wn = wave & 3;  // 2 x 4 wave grid
  const int lm = lane & 15, lh = lane >> 4;
  const int m0 = blockIdx.y * BM;
  const int n0 = blockIdx.x * BN;

  v8f acc[4][2];
#pragma unroll
  for (int i = 0; i < 4; ++i)
#pragma unroll
    for (int j = 0; j < 2; ++j) acc[i][j] = (v8f){};

  // prologue: queue tile 0 into buffer 0
  if (wave == 0) {
    tdm_load_2d(lds_off(&sA[0][0][0]), A + (long)m0 * K, BK, BM, K, M, K);
    tdm_load_2d(lds_off(&sB[0][0][0]), Bt + (long)n0 * K, BK, BN, K, N, K);
  }

  int cur = 0;
  for (int kt = 0; kt < K; kt += BK) {
    if (wave == 0) {
      if (kt + BK < K) {  // queue next tile into the other buffer
        tdm_load_2d(lds_off(&sA[cur ^ 1][0][0]), A + (long)m0 * K + kt + BK,
                    BK, BM, K, M, K);
        tdm_load_2d(lds_off(&sB[cur ^ 1][0][0]), Bt + (long)n0 * K + kt + BK,
                    BK, BN, K, N, K);
        TDM_WAIT(2);  // current tile's 2 loads complete (in-order)
      } else {
        TDM_WAIT(0);
      }
    }
    __syncthreads();

#pragma unroll
    for (int kc = 0; kc < 2; ++kc) {  // two K=32 WMMA chunks per tile
      v16bf af[4], bfr[2];
#pragma unroll
      for (int i = 0; i < 4; ++i) {  // A-frag: lane=m+16h, runs at 8h, 16+8h
        const __bf16* ap = &sA[cur][wm * 64 + i * 16 + lm][kc * 32];
        af[i] = cat8(*(const bf16x8*)(ap + 8 * lh),
                     *(const bf16x8*)(ap + 16 + 8 * lh));
      }
#pragma unroll
      for (int j = 0; j < 2; ++j) {  // B-frag: lane=n+16h, runs at 8h, 16+8h
        const __bf16* bp = &sB[cur][wn * 32 + j * 16 + lm][kc * 32];
        bfr[j] = cat8(*(const bf16x8*)(bp + 8 * lh),
                      *(const bf16x8*)(bp + 16 + 8 * lh));
      }
#pragma unroll
      for (int i = 0; i < 4; ++i)
#pragma unroll
        for (int j = 0; j < 2; ++j)
          acc[i][j] = wmma_bf16(af[i], bfr[j], acc[i][j]);
    }
    __syncthreads();  // done reading `cur`; TDM may overwrite it next round
    cur ^= 1;
  }

  // Epilogue. C layout: vgpr r -> row r + 8*(lane/16), col lane%16.
#pragma unroll
  for (int i = 0; i < 4; ++i) {
#pragma unroll
    for (int j = 0; j < 2; ++j) {
      const int col = n0 + wn * 32 + j * 16 + lm;
      const float bv = bias[col];
#pragma unroll
      for (int r = 0; r < 8; ++r) {
        const int row = m0 + wm * 64 + i * 16 + r + 8 * lh;
        float v = acc[i][j][r] + bv;
        if (EPI == 2) v = 0.5f * v * (1.0f + erff(v * 0.70710678118654752f));
        if (EPI == 0) Cf[(long)row * N + col] = v;
        else          Cb[(long)row * N + col] = (__bf16)v;
      }
    }
  }
}

// ---------------------------------------------------------------------------
// Kernel 4: flash attention. Block = (qblock of 64 rows, head, batch),
// 128 threads = 4 waves; each wave owns 16 query rows. K/V tiles (32 keys)
// staged in LDS, online softmax with clipped rel-pos bias.
// qkv layout: [BSEQ][3*D_MODEL] bf16, cols: Q=h*64, K=1024+h*64, V=2048+h*64
// ---------------------------------------------------------------------------
__global__ __launch_bounds__(128)
void attention_kernel(const __bf16* __restrict__ qkv,
                      const float* __restrict__ relb,
                      __bf16* __restrict__ out) {
  __shared__ __align__(16) float  sRel[128];
  __shared__ __align__(16) __bf16 sK[32][72];      // [key][d]
  __shared__ __align__(16) __bf16 sV[64][40];      // [d][key] (transposed)
  __shared__ __align__(16) __bf16 sP[4][16][32];   // per-wave probs

  const int qblk = blockIdx.x, head = blockIdx.y, batch = blockIdx.z;
  const int tid = threadIdx.x, wave = tid >> 5, lane = tid & 31;
  const int lm = lane & 15, lh = lane >> 4;

  if (tid < 127) sRel[tid] = relb[tid];

  const long rowStride = 3 * D_MODEL;
  const long base = (long)batch * SEQ * rowStride;
  const int qcol = head * HEAD_DIM;
  const int kcol = D_MODEL + head * HEAD_DIM;
  const int vcol = 2 * D_MODEL + head * HEAD_DIM;

  // Q A-fragments (16 rows x 64 d = two K=32 chunks), loaded once.
  const int qrow0 = qblk * 64 + wave * 16;
  v16bf aq[2];
  {
    const __bf16* qp = qkv + base + (long)(qrow0 + lm) * rowStride + qcol;
#pragma unroll
    for (int c = 0; c < 2; ++c)
      aq[c] = cat8(*(const bf16x8*)(qp + c * 32 + 8 * lh),
                   *(const bf16x8*)(qp + c * 32 + 16 + 8 * lh));
  }

  float mrun[8], lrun[8];
  v8f o[4];
#pragma unroll
  for (int r = 0; r < 8; ++r) { mrun[r] = -1e30f; lrun[r] = 0.0f; }
#pragma unroll
  for (int d = 0; d < 4; ++d) o[d] = (v8f){};

  const int skey = tid >> 2;        // staging: key 0..31
  const int sd = (tid & 3) * 16;    // staging: d offset 0/16/32/48

  for (int kb = 0; kb < SEQ; kb += 32) {
    {  // stage K row-major, V transposed
      const __bf16* kp = qkv + base + (long)(kb + skey) * rowStride + kcol + sd;
      *(uint4*)&sK[skey][sd]     = *(const uint4*)kp;
      *(uint4*)&sK[skey][sd + 8] = *(const uint4*)(kp + 8);
      const __bf16* vp = qkv + base + (long)(kb + skey) * rowStride + vcol + sd;
      bf16x8 v0 = *(const bf16x8*)vp;
      bf16x8 v1 = *(const bf16x8*)(vp + 8);
#pragma unroll
      for (int e = 0; e < 8; ++e) {
        sV[sd + e][skey]     = v0[e];
        sV[sd + 8 + e][skey] = v1[e];
      }
    }
    __syncthreads();

    // scores: 16 q rows x 32 keys (two 16-key subtiles, K-dim 64 = 2 WMMAs)
    v8f s[2];
#pragma unroll
    for (int ns = 0; ns < 2; ++ns) {
      v8f c = (v8f){};
#pragma unroll
      for (int dc = 0; dc < 2; ++dc) {
        const __bf16* bp = &sK[ns * 16 + lm][dc * 32];
        c = wmma_bf16(aq[dc],
                      cat8(*(const bf16x8*)(bp + 8 * lh),
                           *(const bf16x8*)(bp + 16 + 8 * lh)),
                      c);
      }
      s[ns] = c;
    }

    // scale + rel-pos bias + online softmax (per-row over 16 lanes of a half)
#pragma unroll
    for (int r = 0; r < 8; ++r) {
      const int qrow = qrow0 + r + 8 * lh;
      float v0 = s[0][r] * 0.125f;
      float v1 = s[1][r] * 0.125f;
      int r0 = qrow - (kb + lm) + 63;      r0 = r0 < 0 ? 0 : (r0 > 126 ? 126 : r0);
      int r1 = qrow - (kb + 16 + lm) + 63; r1 = r1 < 0 ? 0 : (r1 > 126 ? 126 : r1);
      v0 += sRel[r0];
      v1 += sRel[r1];
      float mx = fmaxf(v0, v1);
#pragma unroll
      for (int off = 1; off < 16; off <<= 1)
        mx = fmaxf(mx, __shfl_xor(mx, off, 16));
      const float mnew = fmaxf(mrun[r], mx);
      const float alpha = __expf(mrun[r] - mnew);
      const float p0 = __expf(v0 - mnew);
      const float p1 = __expf(v1 - mnew);
      float ps = p0 + p1;
#pragma unroll
      for (int off = 1; off < 16; off <<= 1)
        ps += __shfl_xor(ps, off, 16);
      lrun[r] = lrun[r] * alpha + ps;
      mrun[r] = mnew;
#pragma unroll
      for (int d = 0; d < 4; ++d) o[d][r] = o[d][r] * alpha;
      const int prow = r + 8 * lh;
      sP[wave][prow][lm]      = (__bf16)p0;
      sP[wave][prow][16 + lm] = (__bf16)p1;
    }
    __syncthreads();

    // P(16x32) @ V(32x64): 4 WMMAs over d-tiles
    v16bf ap;
    {
      const __bf16* pp = &sP[wave][lm][0];
      ap = cat8(*(const bf16x8*)(pp + 8 * lh),
                *(const bf16x8*)(pp + 16 + 8 * lh));
    }
#pragma unroll
    for (int d = 0; d < 4; ++d) {
      const __bf16* vp2 = &sV[d * 16 + lm][0];
      o[d] = wmma_bf16(ap,
                       cat8(*(const bf16x8*)(vp2 + 8 * lh),
                            *(const bf16x8*)(vp2 + 16 + 8 * lh)),
                       o[d]);
    }
    __syncthreads();
  }

  // normalize and write bf16 attn output [BSEQ][D_MODEL]
#pragma unroll
  for (int d = 0; d < 4; ++d)
#pragma unroll
    for (int r = 0; r < 8; ++r) {
      const int qrow = qrow0 + r + 8 * lh;
      const float v = o[d][r] / lrun[r];
      out[((long)batch * SEQ + qrow) * D_MODEL + head * HEAD_DIM + d * 16 + lm] =
          (__bf16)v;
    }
}

// ---------------------------------------------------------------------------
// Kernel 5: y = LayerNorm(a + b) * g + beta ; optional bf16 copy of y
// One block per row (D_MODEL = 1024, 256 threads x 4 elems).
// ---------------------------------------------------------------------------
__global__ __launch_bounds__(256)
void add_ln_kernel(const float* __restrict__ a, const float* __restrict__ bsum,
                   const float* __restrict__ g, const float* __restrict__ beta,
                   float* __restrict__ of, __bf16* __restrict__ ob) {
  __shared__ float red[256];
  const int row = blockIdx.x, tid = threadIdx.x;
  const long base = (long)row * D_MODEL;
  float v[4];
  float s = 0.0f;
#pragma unroll
  for (int i = 0; i < 4; ++i) {
    const int d = tid + i * 256;
    v[i] = a[base + d] + bsum[base + d];
    s += v[i];
  }
  red[tid] = s;
  __syncthreads();
  for (int st = 128; st > 0; st >>= 1) {
    if (tid < st) red[tid] += red[tid + st];
    __syncthreads();
  }
  const float mean = red[0] * (1.0f / D_MODEL);
  __syncthreads();
  s = 0.0f;
#pragma unroll
  for (int i = 0; i < 4; ++i) {
    const float t = v[i] - mean;
    s += t * t;
  }
  red[tid] = s;
  __syncthreads();
  for (int st = 128; st > 0; st >>= 1) {
    if (tid < st) red[tid] += red[tid + st];
    __syncthreads();
  }
  const float rstd = rsqrtf(red[0] * (1.0f / D_MODEL) + 1e-5f);
#pragma unroll
  for (int i = 0; i < 4; ++i) {
    const int d = tid + i * 256;
    const float y = (v[i] - mean) * rstd * g[d] + beta[d];
    of[base + d] = y;
    if (ob) ob[base + d] = (__bf16)y;
  }
}

// ---------------------------------------------------------------------------
// Host orchestration
// ---------------------------------------------------------------------------
extern "C" void kernel_launch(void* const* d_in, const int* in_sizes, int n_in,
                              void* d_out, int out_size, void* d_ws, size_t ws_size,
                              hipStream_t stream) {
  (void)in_sizes; (void)n_in; (void)out_size; (void)ws_size;

  const float* x      = (const float*)d_in[0];
  const float* se     = (const float*)d_in[1];
  const float* qkv_w  = (const float*)d_in[2];
  const float* qkv_b  = (const float*)d_in[3];
  const float* out_w  = (const float*)d_in[4];
  const float* out_b  = (const float*)d_in[5];
  const float* relb   = (const float*)d_in[6];
  const float* w1     = (const float*)d_in[7];
  const float* b1     = (const float*)d_in[8];
  const float* w2     = (const float*)d_in[9];
  const float* b2     = (const float*)d_in[10];
  const float* ln1_g  = (const float*)d_in[11];
  const float* ln1_b  = (const float*)d_in[12];
  const float* ln2_g  = (const float*)d_in[13];
  const float* ln2_b  = (const float*)d_in[14];
  const int*   stepp  = (const int*)d_in[15];

  // workspace carve (256B aligned)
  char* wp = (char*)d_ws;
  auto carve = [&](size_t bytes) {
    char* p = wp;
    wp += (bytes + 255) & ~(size_t)255;
    return p;
  };
  float*  xres  = (float*) carve((size_t)BSEQ * D_MODEL * 4);
  __bf16* xb    = (__bf16*)carve((size_t)BSEQ * D_MODEL * 2);
  __bf16* qkvwt = (__bf16*)carve((size_t)3 * D_MODEL * D_MODEL * 2);
  __bf16* outwt = (__bf16*)carve((size_t)D_MODEL * D_MODEL * 2);
  __bf16* w1t   = (__bf16*)carve((size_t)D_MODEL * D_FF * 2);
  __bf16* w2t   = (__bf16*)carve((size_t)D_FF * D_MODEL * 2);
  __bf16* qkvb  = (__bf16*)carve((size_t)BSEQ * 3 * D_MODEL * 2);
  __bf16* attnb = (__bf16*)carve((size_t)BSEQ * D_MODEL * 2);
  float*  projf = (float*) carve((size_t)BSEQ * D_MODEL * 4);
  float*  h1f   = (float*) carve((size_t)BSEQ * D_MODEL * 4);
  __bf16* h1b   = (__bf16*)carve((size_t)BSEQ * D_MODEL * 2);
  __bf16* ffn1b = (__bf16*)carve((size_t)BSEQ * D_FF * 2);
  float*  ffn2f = (float*) carve((size_t)BSEQ * D_MODEL * 4);

  // 1) step-embed add + bf16 convert
  {
    const int n = BSEQ * D_MODEL;
    prep_kernel<<<(n + 255) / 256, 256, 0, stream>>>(x, se, stepp, xres, xb, n);
  }
  // 2) weight transpose-convert to bf16 [N][K]
  {
    long t;
    t = (long)D_MODEL * 3 * D_MODEL;
    wt_convert_kernel<<<(t + 255) / 256, 256, 0, stream>>>(qkv_w, qkvwt, D_MODEL, 3 * D_MODEL);
    t = (long)D_MODEL * D_MODEL;
    wt_convert_kernel<<<(t + 255) / 256, 256, 0, stream>>>(out_w, outwt, D_MODEL, D_MODEL);
    t = (long)D_MODEL * D_FF;
    wt_convert_kernel<<<(t + 255) / 256, 256, 0, stream>>>(w1, w1t, D_MODEL, D_FF);
    t = (long)D_FF * D_MODEL;
    wt_convert_kernel<<<(t + 255) / 256, 256, 0, stream>>>(w2, w2t, D_FF, D_MODEL);
  }
  // 3) QKV projection -> bf16 [BSEQ][3072]
  gemm_bf16_kernel<1><<<dim3(3 * D_MODEL / 128, BSEQ / 128), 256, 0, stream>>>(
      xb, qkvwt, qkv_b, nullptr, qkvb, BSEQ, 3 * D_MODEL, D_MODEL);
  // 4) attention -> bf16 [BSEQ][1024]
  attention_kernel<<<dim3(SEQ / 64, N_HEADS, BATCH), 128, 0, stream>>>(qkvb, relb, attnb);
  // 5) output projection -> f32
  gemm_bf16_kernel<0><<<dim3(D_MODEL / 128, BSEQ / 128), 256, 0, stream>>>(
      attnb, outwt, out_b, projf, nullptr, BSEQ, D_MODEL, D_MODEL);
  // 6) LN1(x + attn) -> f32 residual + bf16 operand
  add_ln_kernel<<<BSEQ, 256, 0, stream>>>(xres, projf, ln1_g, ln1_b, h1f, h1b);
  // 7) FFN1 with exact GELU -> bf16 [BSEQ][4096]
  gemm_bf16_kernel<2><<<dim3(D_FF / 128, BSEQ / 128), 256, 0, stream>>>(
      h1b, w1t, b1, nullptr, ffn1b, BSEQ, D_FF, D_MODEL);
  // 8) FFN2 -> f32
  gemm_bf16_kernel<0><<<dim3(D_MODEL / 128, BSEQ / 128), 256, 0, stream>>>(
      ffn1b, w2t, b2, ffn2f, nullptr, BSEQ, D_MODEL, D_FF);
  // 9) LN2(h1 + ffn) -> final f32 output
  add_ln_kernel<<<BSEQ, 256, 0, stream>>>(h1f, ffn2f, ln2_g, ln2_b, (float*)d_out, nullptr);
}
